// LSTM_11708080849329
// MI455X (gfx1250) — matrix operand order repaired and checked by reading, not compiled, not run on hardware
//
#include <hip/hip_runtime.h>
#include <hip/hip_bf16.h>

// ---------------------------------------------------------------------------
// 2-layer LSTM (B=64, T=256, I=256, H=1024) + projection H->1000 on gfx1250.
// Persistent kernel: 16 blocks x 256 threads resident for all 256 timesteps;
// device-side grid barrier (agent atomics + s_sleep). Each wave owns
// (gate, 32-col slice, all 64 rows): per k-step 6 fragment loads (2 B + 4 A)
// feed 8 v_wmma_f32_16x16x32_bf16 -> 0.75 KB operand fill per WMMA, and every
// bf16 weight byte is read from L2 exactly once per timestep. Gates are
// recombined via LDS; cell state c lives entirely in registers.
// ---------------------------------------------------------------------------

typedef __attribute__((ext_vector_type(16))) __bf16 v16bf;
typedef __attribute__((ext_vector_type(8)))  __bf16 v8bf;
typedef __attribute__((ext_vector_type(8)))  float  v8f;

#define LSTM_B 64
#define LSTM_T 256
#define LSTM_I 256
#define LSTM_H 1024
#define LSTM_C 1000
#define LSTM_CP 1024
#define NBLOCKS 16

#define WMMA_BF16(a, b, c) \
  __builtin_amdgcn_wmma_f32_16x16x32_bf16(false, (a), false, (b), (short)0, (c), false, false)

// ---- WMMA fragment loaders -------------------------------------------------
// A fragment 16x32 (MxK): lanes 0-15: M=lane, K k0..k0+15;
// lanes 16-31: K k0+16..k0+31. One 32B load.
__device__ inline v16bf load_a_frag(const __bf16* __restrict__ A, int kstride,
                                    int m0, int k0, int lane) {
  const int hlf = lane >> 4;
  const int row = m0 + (lane & 15);
  const __bf16* p = A + (size_t)row * kstride + k0 + hlf * 16;
  return *(const v16bf*)p;
}

// B fragment 32x16 (KxN) from N-major weights Wt[n][k]: lanes 0-15: N=n0+lane,
// K k0..k0+15; lanes 16-31: N=n0+lane-16, K k0+16..k0+31. One 32B load.
__device__ inline v16bf load_b_frag(const __bf16* __restrict__ Wt, int kstride,
                                    int n0, int k0, int lane) {
  const int hlf = lane >> 4;
  const int col = n0 + (lane & 15);
  const __bf16* p = Wt + (size_t)col * kstride + k0 + hlf * 16;
  return *(const v16bf*)p;
}

__device__ inline float sigmoidf_fast(float x) {
  return 1.0f / (1.0f + __expf(-x));
}

// ---- prep kernels ----------------------------------------------------------
// W [g][k][n] fp32 -> Wt [g][n][k] bf16, n >= N zero-padded to Npad.
__global__ __launch_bounds__(256)
void conv_transpose_w(const float* __restrict__ W, __bf16* __restrict__ Wt,
                      int G, int K, int N, int Npad) {
  size_t i = (size_t)blockIdx.x * blockDim.x + threadIdx.x;
  size_t total = (size_t)G * Npad * K;
  if (i >= total) return;
  int k = (int)(i % K);
  int n = (int)((i / K) % Npad);
  int g = (int)(i / ((size_t)K * Npad));
  float v = (n < N) ? W[((size_t)g * K + k) * N + n] : 0.0f;
  Wt[i] = (__bf16)v;
}

// x [b][t][i] fp32 -> xbf [t][b][i] bf16
__global__ __launch_bounds__(256)
void conv_x(const float* __restrict__ x, __bf16* __restrict__ xbf) {
  size_t i = (size_t)blockIdx.x * blockDim.x + threadIdx.x;
  if (i >= (size_t)LSTM_B * LSTM_T * LSTM_I) return;
  int ii = (int)(i % LSTM_I);
  int t  = (int)((i / LSTM_I) % LSTM_T);
  int b  = (int)(i / ((size_t)LSTM_I * LSTM_T));
  xbf[((size_t)t * LSTM_B + b) * LSTM_I + ii] = (__bf16)x[i];
}

__global__ __launch_bounds__(256)
void init_state(__bf16* h0, __bf16* h1, unsigned* barrier_vars) {
  int i = blockIdx.x * blockDim.x + threadIdx.x;
  if (i < LSTM_B * LSTM_H) { h0[i] = (__bf16)0.0f; h1[i] = (__bf16)0.0f; }
  if (i < 2) barrier_vars[i] = 0u;
}

// ---- persistent LSTM kernel ------------------------------------------------
__global__ __launch_bounds__(256)
void lstm_persistent(const __bf16* __restrict__ xbf,
                     const __bf16* __restrict__ W0x_t, const __bf16* __restrict__ W0h_t,
                     const float*  __restrict__ b0,
                     const __bf16* __restrict__ W1x_t, const __bf16* __restrict__ W1h_t,
                     const float*  __restrict__ b1,
                     const __bf16* __restrict__ Wo_t,  const float* __restrict__ bo_,
                     __bf16* h0a, __bf16* h0b, __bf16* h1a, __bf16* h1b,
                     float* __restrict__ out,
                     unsigned* bar_counter, unsigned* bar_gen)
{
  const int H = LSTM_H;
  // z staging: [n-pair-in-block][gate][row][col 0..31], 64 KB
  __shared__ float zbuf[2][4][64][32];

  const int wave = threadIdx.x >> 5;
  const int lane = threadIdx.x & 31;
  const int hlf  = lane >> 4;
  const int npair = wave >> 2;                          // 0..1 (32-col group)
  const int gate  = wave & 3;                           // 0..3
  const int n0    = blockIdx.x * 64 + npair * 32;       // this wave: cols n0..n0+31

  // cell state in registers: thread i statically owns the same 16+16 cells
  // every timestep (eid = i*256 + tid mapping below).
  float c0reg[16], c1reg[16];
#pragma unroll
  for (int i = 0; i < 16; ++i) { c0reg[i] = 0.0f; c1reg[i] = 0.0f; }

  // ---- grid barrier (generation-counted; deterministic across replays) ----
  auto grid_sync = [&]() {
    __threadfence();
    __syncthreads();
    if (threadIdx.x == 0) {
      unsigned old = __hip_atomic_load(bar_gen, __ATOMIC_ACQUIRE, __HIP_MEMORY_SCOPE_AGENT);
      unsigned arrived = __hip_atomic_fetch_add(bar_counter, 1u, __ATOMIC_ACQ_REL,
                                                __HIP_MEMORY_SCOPE_AGENT);
      if (arrived == NBLOCKS - 1) {
        __hip_atomic_store(bar_counter, 0u, __ATOMIC_RELAXED, __HIP_MEMORY_SCOPE_AGENT);
        __hip_atomic_fetch_add(bar_gen, 1u, __ATOMIC_RELEASE, __HIP_MEMORY_SCOPE_AGENT);
      } else {
        while (__hip_atomic_load(bar_gen, __ATOMIC_ACQUIRE, __HIP_MEMORY_SCOPE_AGENT) == old)
          __builtin_amdgcn_s_sleep(2);
      }
    }
    __syncthreads();
  };

  // ---- one layer, one timestep ----
  // This wave: gate `gate`, cols n0..n0+31 (2 n-tiles), all 64 rows (4 m-tiles).
  // 8 accumulators; per k-step 2 B + 4 A fragment loads feed 8 WMMAs.
  auto layer_step = [&](const __bf16* Ax, int Kx, const __bf16* Hprev,
                        const __bf16* Wxt, const __bf16* Wht,
                        const float* bias, float* creg, __bf16* Hout) {
    v8f acc[4][2] = {};
    const __bf16* Wgx = Wxt + (size_t)gate * H * Kx;
    const __bf16* Wgh = Wht + (size_t)gate * H * H;

    for (int k0 = 0; k0 < Kx; k0 += 32) {
      v16bf bA = load_b_frag(Wgx, Kx, n0,      k0, lane);
      v16bf bB = load_b_frag(Wgx, Kx, n0 + 16, k0, lane);
      v16bf a0 = load_a_frag(Ax, Kx,  0, k0, lane);
      v16bf a1 = load_a_frag(Ax, Kx, 16, k0, lane);
      v16bf a2 = load_a_frag(Ax, Kx, 32, k0, lane);
      v16bf a3 = load_a_frag(Ax, Kx, 48, k0, lane);
      acc[0][0] = WMMA_BF16(a0, bA, acc[0][0]);
      acc[1][0] = WMMA_BF16(a1, bA, acc[1][0]);
      acc[2][0] = WMMA_BF16(a2, bA, acc[2][0]);
      acc[3][0] = WMMA_BF16(a3, bA, acc[3][0]);
      acc[0][1] = WMMA_BF16(a0, bB, acc[0][1]);
      acc[1][1] = WMMA_BF16(a1, bB, acc[1][1]);
      acc[2][1] = WMMA_BF16(a2, bB, acc[2][1]);
      acc[3][1] = WMMA_BF16(a3, bB, acc[3][1]);
    }
    for (int k0 = 0; k0 < H; k0 += 32) {
      v16bf bA = load_b_frag(Wgh, H, n0,      k0, lane);
      v16bf bB = load_b_frag(Wgh, H, n0 + 16, k0, lane);
      v16bf a0 = load_a_frag(Hprev, H,  0, k0, lane);
      v16bf a1 = load_a_frag(Hprev, H, 16, k0, lane);
      v16bf a2 = load_a_frag(Hprev, H, 32, k0, lane);
      v16bf a3 = load_a_frag(Hprev, H, 48, k0, lane);
      acc[0][0] = WMMA_BF16(a0, bA, acc[0][0]);
      acc[1][0] = WMMA_BF16(a1, bA, acc[1][0]);
      acc[2][0] = WMMA_BF16(a2, bA, acc[2][0]);
      acc[3][0] = WMMA_BF16(a3, bA, acc[3][0]);
      acc[0][1] = WMMA_BF16(a0, bB, acc[0][1]);
      acc[1][1] = WMMA_BF16(a1, bB, acc[1][1]);
      acc[2][1] = WMMA_BF16(a2, bB, acc[2][1]);
      acc[3][1] = WMMA_BF16(a3, bB, acc[3][1]);
    }

    // stage z to LDS. C/D layout: elem j -> row j+8*(lane>=16), col lane&15.
    const int col = lane & 15;
#pragma unroll
    for (int m = 0; m < 4; ++m) {
#pragma unroll
      for (int j = 0; j < 8; ++j) {
        zbuf[npair][gate][m * 16 + j + hlf * 8][col]      = acc[m][0][j];
        zbuf[npair][gate][m * 16 + j + hlf * 8][col + 16] = acc[m][1][j];
      }
    }
    __syncthreads();

    // fused gate combine + cell update: 4096 elems / 256 threads = 16 each.
#pragma unroll
    for (int i = 0; i < 16; ++i) {
      const int eid = i * 256 + threadIdx.x;
      const int p   = eid >> 11;            // n-pair in block
      const int row = (eid >> 5) & 63;
      const int cc  = eid & 31;
      const int gcol = blockIdx.x * 64 + p * 32 + cc;
      const float zg = zbuf[p][0][row][cc] + bias[0 * H + gcol];
      const float zi = zbuf[p][1][row][cc] + bias[1 * H + gcol];
      const float zf = zbuf[p][2][row][cc] + bias[2 * H + gcol];
      const float zo = zbuf[p][3][row][cc] + bias[3 * H + gcol];
      const float cn = tanhf(zg) * sigmoidf_fast(zi) + creg[i] * sigmoidf_fast(zf);
      creg[i] = cn;
      Hout[(size_t)row * H + gcol] = (__bf16)(tanhf(cn) * sigmoidf_fast(zo));
    }
    // grid_sync() after return provides the LDS/global ordering for next phase
  };

  // ---- recurrence ----
  for (int t = 0; t < LSTM_T; ++t) {
    const __bf16* h0_in  = (t & 1) ? h0b : h0a;
    __bf16*       h0_out = (t & 1) ? h0a : h0b;
    const __bf16* h1_in  = (t & 1) ? h1b : h1a;
    __bf16*       h1_out = (t & 1) ? h1a : h1b;

    layer_step(xbf + (size_t)t * LSTM_B * LSTM_I, LSTM_I, h0_in,
               W0x_t, W0h_t, b0, c0reg, h0_out);
    grid_sync();
    layer_step(h0_out, LSTM_H, h1_in, W1x_t, W1h_t, b1, c1reg, h1_out);
    grid_sync();
  }

  // ---- fused projection: out[64,1000] = h_final @ Wo + bo ----
  const __bf16* hfin = ((LSTM_T - 1) & 1) ? h1a : h1b;  // T=256 -> h1a
#pragma unroll
  for (int rep = 0; rep < 2; ++rep) {
    const int gw = rep * 128 + blockIdx.x * 8 + wave;   // 256 tiles / 128 waves
    const int m0 = (gw >> 6) << 4;
    const int n0p = (gw & 63) << 4;
    v8f acc = {};
    for (int k0 = 0; k0 < H; k0 += 32) {
      v16bf a = load_a_frag(hfin, H, m0, k0, lane);
      v16bf b = load_b_frag(Wo_t, H, n0p, k0, lane);
      acc = WMMA_BF16(a, b, acc);
    }
    const int col = n0p + (lane & 15);
    if (col < LSTM_C) {
      const float bv = bo_[col];
#pragma unroll
      for (int j = 0; j < 8; ++j) {
        const int row = m0 + j + hlf * 8;
        out[(size_t)row * LSTM_C + col] = acc[j] + bv;
      }
    }
  }
}

// ---------------------------------------------------------------------------
extern "C" void kernel_launch(void* const* d_in, const int* in_sizes, int n_in,
                              void* d_out, int out_size, void* d_ws, size_t ws_size,
                              hipStream_t stream) {
  const float* x   = (const float*)d_in[0];  // [64,256,256]
  const float* W0x = (const float*)d_in[1];  // [4,256,1024]
  const float* W0h = (const float*)d_in[2];  // [4,1024,1024]
  const float* b0  = (const float*)d_in[3];  // [4,1024]
  const float* W1x = (const float*)d_in[4];  // [4,1024,1024]
  const float* W1h = (const float*)d_in[5];  // [4,1024,1024]
  const float* b1  = (const float*)d_in[6];  // [4,1024]
  const float* Wo  = (const float*)d_in[7];  // [1024,1000]
  const float* bo  = (const float*)d_in[8];  // [1000]
  float* out = (float*)d_out;                // [64,1000]
  (void)in_sizes; (void)n_in; (void)out_size; (void)ws_size;

  // ---- carve workspace (all chunks multiples of 256B) ----
  char* ws = (char*)d_ws;
  size_t off = 0;
  auto carve = [&](size_t bytes) { char* p = ws + off; off += bytes; return p; };

  unsigned* bar   = (unsigned*)carve(256);                             // counter, gen
  __bf16* W0x_t = (__bf16*)carve((size_t)4 * LSTM_H * LSTM_I * 2);     // [4][1024][256]
  __bf16* W0h_t = (__bf16*)carve((size_t)4 * LSTM_H * LSTM_H * 2);     // [4][1024][1024]
  __bf16* W1x_t = (__bf16*)carve((size_t)4 * LSTM_H * LSTM_H * 2);
  __bf16* W1h_t = (__bf16*)carve((size_t)4 * LSTM_H * LSTM_H * 2);
  __bf16* Wo_t  = (__bf16*)carve((size_t)LSTM_CP * LSTM_H * 2);        // padded
  __bf16* xbf   = (__bf16*)carve((size_t)LSTM_T * LSTM_B * LSTM_I * 2);
  __bf16* h0a   = (__bf16*)carve((size_t)LSTM_B * LSTM_H * 2);
  __bf16* h0b   = (__bf16*)carve((size_t)LSTM_B * LSTM_H * 2);
  __bf16* h1a   = (__bf16*)carve((size_t)LSTM_B * LSTM_H * 2);
  __bf16* h1b   = (__bf16*)carve((size_t)LSTM_B * LSTM_H * 2);

  // ---- per-call prep: bf16 transposed weights, bf16 x, zeroed state ----
  size_t t;
  t = (size_t)4 * LSTM_H * LSTM_I;
  conv_transpose_w<<<(unsigned)((t + 255) / 256), 256, 0, stream>>>(W0x, W0x_t, 4, LSTM_I, LSTM_H, LSTM_H);
  t = (size_t)4 * LSTM_H * LSTM_H;
  conv_transpose_w<<<(unsigned)((t + 255) / 256), 256, 0, stream>>>(W0h, W0h_t, 4, LSTM_H, LSTM_H, LSTM_H);
  conv_transpose_w<<<(unsigned)((t + 255) / 256), 256, 0, stream>>>(W1x, W1x_t, 4, LSTM_H, LSTM_H, LSTM_H);
  conv_transpose_w<<<(unsigned)((t + 255) / 256), 256, 0, stream>>>(W1h, W1h_t, 4, LSTM_H, LSTM_H, LSTM_H);
  t = (size_t)LSTM_CP * LSTM_H;
  conv_transpose_w<<<(unsigned)((t + 255) / 256), 256, 0, stream>>>(Wo, Wo_t, 1, LSTM_H, LSTM_C, LSTM_CP);
  t = (size_t)LSTM_B * LSTM_T * LSTM_I;
  conv_x<<<(unsigned)((t + 255) / 256), 256, 0, stream>>>(x, xbf);
  init_state<<<(LSTM_B * LSTM_H) / 256, 256, 0, stream>>>(h0a, h1a, bar);

  // ---- single persistent kernel runs the whole recurrence + projection ----
  lstm_persistent<<<NBLOCKS, 256, 0, stream>>>(
      xbf, W0x_t, W0h_t, b0, W1x_t, W1h_t, b1, Wo_t, bo,
      h0a, h0b, h1a, h1b, out, bar + 0, bar + 1);
}
